// SRDecoder_9835475108279
// MI455X (gfx1250) — matrix operand order repaired and checked by reading, not compiled, not run on hardware
//
#include <hip/hip_runtime.h>

// ---------------- problem constants ----------------
constexpr int kB  = 256;   // batch
constexpr int kT  = 256;   // timesteps
constexpr int kE  = 256;   // embedding dim
constexpr int kH  = 1024;  // hidden
constexpr int kV  = 1024;  // vocab
constexpr int kZD = 256;   // z dim
constexpr int kKx0 = 2 * kE; // 512, layer0 input width ([prev_emb | z_emb])

// ---------------- WMMA plumbing ----------------
typedef __bf16 v16bf __attribute__((ext_vector_type(16)));
typedef float  v8f   __attribute__((ext_vector_type(8)));

union FragBF {
  v16bf v;
  uint4 q[2];
};

__device__ inline v8f wmma_bf16(v16bf a, v16bf b, v8f c) {
  return __builtin_amdgcn_wmma_f32_16x16x32_bf16(
      /*neg_a=*/false, a, /*neg_b=*/false, b,
      /*c_mod=*/(short)0, c, /*reuse_a=*/false, /*reuse_b=*/false);
}

// A fragment: 16x32 (MxK) bf16 from row-major X (ld = row length).
// lane l (l<16): row M=l, K = {k0..k0+7} in V0-3, {k0+16..k0+23} in V4-7
// lane l+16   : row M=l, K = {k0+8..}, {k0+24..}
__device__ inline v16bf load_a_frag(const __bf16* __restrict__ X, int ld,
                                    int mbase, int k0, int lane) {
  const int m  = lane & 15;
  const int kh = (lane >> 4) << 3;                 // 0 or 8
  const __bf16* base = X + (size_t)(mbase + m) * ld + k0 + kh;
  FragBF f;
  f.q[0] = *reinterpret_cast<const uint4*>(base);        // 8 bf16
  f.q[1] = *reinterpret_cast<const uint4*>(base + 16);   // 8 bf16
  return f.v;
}

// B fragment: 32x16 (KxN) where B[k][n] = W[nbase+n][k], W row-major (out x in).
// lane l (l<16): col N=l, K = {k0..k0+7}, {k0+16..k0+23}
// lane l+16   : col N=l, K = {k0+8..k0+15}, {k0+24..k0+31}
__device__ inline v16bf load_b_frag(const __bf16* __restrict__ W, int ld,
                                    int nbase, int k0, int lane) {
  const int n  = lane & 15;
  const int kh = (lane >> 4) << 3;                 // 0 or 8
  const __bf16* base = W + (size_t)(nbase + n) * ld + k0 + kh;
  FragBF f;
  f.q[0] = *reinterpret_cast<const uint4*>(base);
  f.q[1] = *reinterpret_cast<const uint4*>(base + 16);
  return f.v;
}

// ---------------- math helpers ----------------
__device__ inline float sigmoidf_fast(float x) { return 1.0f / (1.0f + __expf(-x)); }
__device__ inline float tanhf_fast(float x) {
  float e = __expf(-2.0f * fabsf(x));
  float t = (1.0f - e) / (1.0f + e);
  return x >= 0.0f ? t : -t;
}
__device__ inline float seluf(float x) {
  const float a = 1.6732632423543772f, s = 1.0507009873554805f;
  return x > 0.0f ? s * x : s * a * (__expf(x) - 1.0f);
}

// ---------------- prep kernels ----------------
__global__ void f32_to_bf16_kernel(const float* __restrict__ src,
                                   __bf16* __restrict__ dst, int n) {
  int i = blockIdx.x * blockDim.x + threadIdx.x;
  if (i < n) dst[i] = (__bf16)src[i];
}

// hid = selu(z @ w_z1.T + b_z1)   (B x H)
__global__ void zemb1_kernel(const float* __restrict__ z,
                             const float* __restrict__ w_z1,
                             const float* __restrict__ b_z1,
                             float* __restrict__ hid) {
  int i = blockIdx.x * blockDim.x + threadIdx.x;  // over B*H
  int b = i / kH, j = i % kH;
  float acc = b_z1[j];
  const float* zr = z + (size_t)b * kZD;
  const float* wr = w_z1 + (size_t)j * kZD;
  for (int k = 0; k < kZD; ++k) acc = fmaf(zr[k], wr[k], acc);
  hid[i] = seluf(acc);
}

// z_emb = hid @ w_z2.T + b_z2 -> xbuf[:, E:2E] (bf16)
__global__ void zemb2_kernel(const float* __restrict__ hid,
                             const float* __restrict__ w_z2,
                             const float* __restrict__ b_z2,
                             __bf16* __restrict__ xbuf) {
  int i = blockIdx.x * blockDim.x + threadIdx.x;  // over B*E
  int b = i / kE, e = i % kE;
  float acc = b_z2[e];
  const float* hr = hid + (size_t)b * kH;
  const float* wr = w_z2 + (size_t)e * kH;
  for (int k = 0; k < kH; ++k) acc = fmaf(hr[k], wr[k], acc);
  xbuf[(size_t)b * kKx0 + kE + e] = (__bf16)acc;
}

// zero initial hidden state (f32 + bf16) and seed xbuf[:, 0:E] with x0
__global__ void init_state_kernel(float* __restrict__ h0f, __bf16* __restrict__ h0b,
                                  float* __restrict__ h1f, __bf16* __restrict__ h1b,
                                  const float* __restrict__ x0,
                                  __bf16* __restrict__ xbuf) {
  int i = blockIdx.x * blockDim.x + threadIdx.x;  // over B*H
  h0f[i] = 0.0f; h1f[i] = 0.0f;
  h0b[i] = (__bf16)0.0f; h1b[i] = (__bf16)0.0f;
  if (i < kB * kE) {
    int b = i / kE, e = i % kE;
    xbuf[(size_t)b * kKx0 + e] = (__bf16)x0[e];
  }
}

// ---------------- fused GRU layer (WMMA) ----------------
// One wave computes a 16(batch) x 16(hidden) tile of the new hidden state.
// 4 waves/block; grid = (H/64, B/16).
__global__ __launch_bounds__(128)
void gru_layer_kernel(const __bf16* __restrict__ X, int Kx,
                      const float*  __restrict__ hPrevF,
                      const __bf16* __restrict__ hPrevB,
                      const __bf16* __restrict__ Wih,
                      const __bf16* __restrict__ Whh,
                      const float*  __restrict__ bih,
                      const float*  __restrict__ bhh,
                      float*  __restrict__ hOutF,
                      __bf16* __restrict__ hOutB) {
  const int lane  = threadIdx.x & 31;
  const int wid   = threadIdx.x >> 5;
  const int nbase = (blockIdx.x * 4 + wid) * 16;
  const int mbase = blockIdx.y * 16;
  const int ng    = nbase + (lane & 15);   // global hidden index for this lane's column

  v8f accR, accZ, accIN, accHN;
  const float bR  = bih[ng]          + bhh[ng];
  const float bZ  = bih[kH + ng]     + bhh[kH + ng];
  const float bIN = bih[2 * kH + ng];
  const float bHN = bhh[2 * kH + ng];
  for (int i = 0; i < 8; ++i) { accR[i] = bR; accZ[i] = bZ; accIN[i] = bIN; accHN[i] = bHN; }

  // gi = x @ Wih.T  (gate rows: r=[0,H), z=[H,2H), n=[2H,3H))
  for (int k0 = 0; k0 < Kx; k0 += 32) {
    v16bf a = load_a_frag(X, Kx, mbase, k0, lane);
    accR  = wmma_bf16(a, load_b_frag(Wih, Kx, 0 * kH + nbase, k0, lane), accR);
    accZ  = wmma_bf16(a, load_b_frag(Wih, Kx, 1 * kH + nbase, k0, lane), accZ);
    accIN = wmma_bf16(a, load_b_frag(Wih, Kx, 2 * kH + nbase, k0, lane), accIN);
  }
  // gh = h @ Whh.T
  for (int k0 = 0; k0 < kH; k0 += 32) {
    v16bf a = load_a_frag(hPrevB, kH, mbase, k0, lane);
    accR  = wmma_bf16(a, load_b_frag(Whh, kH, 0 * kH + nbase, k0, lane), accR);
    accZ  = wmma_bf16(a, load_b_frag(Whh, kH, 1 * kH + nbase, k0, lane), accZ);
    accHN = wmma_bf16(a, load_b_frag(Whh, kH, 2 * kH + nbase, k0, lane), accHN);
  }

  // fused gate nonlinearity + state update
  const int mtop = (lane >> 4) << 3;  // C/D layout: VGPR i -> row i (lanes 0-15) / 8+i (16-31)
  for (int i = 0; i < 8; ++i) {
    const int m = mbase + mtop + i;
    const float r  = sigmoidf_fast(accR[i]);
    const float z  = sigmoidf_fast(accZ[i]);
    const float nn = tanhf_fast(accIN[i] + r * accHN[i]);
    const float hp = hPrevF[(size_t)m * kH + ng];
    const float hn = (1.0f - z) * nn + z * hp;
    hOutF[(size_t)m * kH + ng] = hn;
    hOutB[(size_t)m * kH + ng] = (__bf16)hn;
  }
}

// ---------------- output projection: probs = relu(h1 @ w_out.T + b_out) ----------------
__global__ __launch_bounds__(128)
void logits_kernel(const __bf16* __restrict__ h1b,
                   const __bf16* __restrict__ Wout,
                   const float*  __restrict__ bout,
                   float* __restrict__ outW, int t) {
  const int lane  = threadIdx.x & 31;
  const int wid   = threadIdx.x >> 5;
  const int nbase = (blockIdx.x * 4 + wid) * 16;
  const int mbase = blockIdx.y * 16;
  const int ng    = nbase + (lane & 15);

  v8f acc;
  const float bv = bout[ng];
  for (int i = 0; i < 8; ++i) acc[i] = bv;

  for (int k0 = 0; k0 < kH; k0 += 32) {
    v16bf a = load_a_frag(h1b, kH, mbase, k0, lane);
    acc = wmma_bf16(a, load_b_frag(Wout, kH, nbase, k0, lane), acc);
  }

  const int mtop = (lane >> 4) << 3;
  for (int i = 0; i < 8; ++i) {
    const int b = mbase + mtop + i;
    const float p = fmaxf(acc[i], 0.0f);
    outW[(size_t)b * kT * kV + (size_t)t * kV + ng] = p;
  }
}

// ---------------- argmax sampling + embedding gather ----------------
__global__ __launch_bounds__(256)
void argmax_gather_kernel(const float* __restrict__ outW, int t,
                          const float* __restrict__ emb,
                          __bf16* __restrict__ xbuf,
                          int* __restrict__ samples) {
  const int b = blockIdx.x;
  const float* row = outW + (size_t)b * kT * kV + (size_t)t * kV;
  float bestv = -INFINITY;
  int   besti = 0;
  for (int v = threadIdx.x; v < kV; v += 256) {
    float p = row[v];
    if (p > bestv) { bestv = p; besti = v; }   // strict > keeps first occurrence per thread
  }
  __shared__ float sv[256];
  __shared__ int   si[256];
  sv[threadIdx.x] = bestv; si[threadIdx.x] = besti;
  __syncthreads();
  for (int s = 128; s > 0; s >>= 1) {
    if (threadIdx.x < (unsigned)s) {
      float v2 = sv[threadIdx.x + s]; int i2 = si[threadIdx.x + s];
      if (v2 > sv[threadIdx.x] || (v2 == sv[threadIdx.x] && i2 < si[threadIdx.x])) {
        sv[threadIdx.x] = v2; si[threadIdx.x] = i2;
      }
    }
    __syncthreads();
  }
  const int idx = si[0];
  // next-step embedding (bf16) into x buffer's first E columns
  xbuf[(size_t)b * kKx0 + threadIdx.x] = (__bf16)emb[(size_t)idx * kE + threadIdx.x];
  if (threadIdx.x == 0) samples[(size_t)b * kT + t] = idx;
}

// ---------------- host orchestration ----------------
extern "C" void kernel_launch(void* const* d_in, const int* in_sizes, int n_in,
                              void* d_out, int out_size, void* d_ws, size_t ws_size,
                              hipStream_t stream) {
  (void)in_sizes; (void)n_in; (void)out_size; (void)ws_size;
  // inputs (setup_inputs order)
  const float* z      = (const float*)d_in[0];
  // d_in[1] score_tensor, d_in[2] train: unused by the inference path
  const float* emb    = (const float*)d_in[3];
  const float* x0     = (const float*)d_in[4];
  const float* w_z1   = (const float*)d_in[5];
  const float* b_z1   = (const float*)d_in[6];
  const float* w_z2   = (const float*)d_in[7];
  const float* b_z2   = (const float*)d_in[8];
  const float* w_ih0  = (const float*)d_in[9];
  const float* w_hh0  = (const float*)d_in[10];
  const float* b_ih0  = (const float*)d_in[11];
  const float* b_hh0  = (const float*)d_in[12];
  const float* w_ih1  = (const float*)d_in[13];
  const float* w_hh1  = (const float*)d_in[14];
  const float* b_ih1  = (const float*)d_in[15];
  const float* b_hh1  = (const float*)d_in[16];
  const float* w_out  = (const float*)d_in[17];
  const float* b_out  = (const float*)d_in[18];

  float* outW    = (float*)d_out;                                   // (B,T,V) f32
  int*   samples = (int*)((float*)d_out + (size_t)kB * kT * kV);    // (B,1,T) i32

  // workspace carve-up (256B aligned slices)
  char* ws = (char*)d_ws;
  size_t off = 0;
  auto take = [&](size_t bytes) -> char* {
    char* p = ws + off;
    off += (bytes + 255) & ~(size_t)255;
    return p;
  };
  __bf16* wih0b = (__bf16*)take((size_t)3 * kH * kKx0 * 2);
  __bf16* whh0b = (__bf16*)take((size_t)3 * kH * kH   * 2);
  __bf16* wih1b = (__bf16*)take((size_t)3 * kH * kH   * 2);
  __bf16* whh1b = (__bf16*)take((size_t)3 * kH * kH   * 2);
  __bf16* woutb = (__bf16*)take((size_t)kV * kH       * 2);
  __bf16* xbuf  = (__bf16*)take((size_t)kB * kKx0     * 2);
  float*  hid   = (float*) take((size_t)kB * kH       * 4);
  float*  h0f[2]; __bf16* h0b[2];
  float*  h1f[2]; __bf16* h1b[2];
  for (int i = 0; i < 2; ++i) {
    h0f[i] = (float*) take((size_t)kB * kH * 4);
    h0b[i] = (__bf16*)take((size_t)kB * kH * 2);
    h1f[i] = (float*) take((size_t)kB * kH * 4);
    h1b[i] = (__bf16*)take((size_t)kB * kH * 2);
  }

  // ---- one-time prep (runs every call; deterministic) ----
  auto cvt = [&](const float* src, __bf16* dst, int n) {
    f32_to_bf16_kernel<<<(n + 255) / 256, 256, 0, stream>>>(src, dst, n);
  };
  cvt(w_ih0, wih0b, 3 * kH * kKx0);
  cvt(w_hh0, whh0b, 3 * kH * kH);
  cvt(w_ih1, wih1b, 3 * kH * kH);
  cvt(w_hh1, whh1b, 3 * kH * kH);
  cvt(w_out, woutb, kV * kH);

  zemb1_kernel<<<(kB * kH) / 256, 256, 0, stream>>>(z, w_z1, b_z1, hid);
  zemb2_kernel<<<(kB * kE) / 256, 256, 0, stream>>>(hid, w_z2, b_z2, xbuf);
  init_state_kernel<<<(kB * kH) / 256, 256, 0, stream>>>(
      h0f[0], h0b[0], h1f[0], h1b[0], x0, xbuf);

  // ---- autoregressive decode ----
  const dim3 gemmGrid(kH / 64, kB / 16);   // 4 waves/block, 16x16 tile/wave
  int cur = 0;
  for (int t = 0; t < kT; ++t) {
    const int nxt = cur ^ 1;
    gru_layer_kernel<<<gemmGrid, 128, 0, stream>>>(
        xbuf, kKx0, h0f[cur], h0b[cur], wih0b, whh0b, b_ih0, b_hh0,
        h0f[nxt], h0b[nxt]);
    gru_layer_kernel<<<gemmGrid, 128, 0, stream>>>(
        h0b[nxt], kH, h1f[cur], h1b[cur], wih1b, whh1b, b_ih1, b_hh1,
        h1f[nxt], h1b[nxt]);
    logits_kernel<<<dim3(kV / 64, kB / 16), 128, 0, stream>>>(
        h1b[nxt], woutb, b_out, outW, t);
    argmax_gather_kernel<<<kB, 256, 0, stream>>>(outW, t, emb, xbuf, samples);
    cur = nxt;
  }
}
